// HybridPillarBlock_16561393893629
// MI455X (gfx1250) — matrix operand order repaired and checked by reading, not compiled, not run on hardware
//
#include <hip/hip_runtime.h>
#include <hip/hip_bf16.h>

// ---------------------------------------------------------------------------
// HybridPillarBlock for MI455X (gfx1250): all GEMMs + attention on WMMA f16,
// async-to-LDS double-buffered tile staging where the toolchain supports it.
// ---------------------------------------------------------------------------

typedef __attribute__((ext_vector_type(16))) _Float16 v16h;
typedef __attribute__((ext_vector_type(8)))  float    v8f;
typedef int v4i_gcc __attribute__((vector_size(16)));   // matches builtin proto

constexpr int Dm  = 1024;   // model dim
constexpr int Hh  = 16;     // heads
constexpr int HDm = 64;     // head dim
constexpr int Bb  = 4;      // batch
constexpr int Ss  = 2048;   // seq
constexpr int NT  = Bb * Ss;    // 8192 tokens
constexpr int DFF = 4 * Dm;     // 4096

#if defined(__has_builtin)
#  if __has_builtin(__builtin_amdgcn_global_load_async_to_lds_b128)
#    define HAVE_ASYNC_LDS 1
#  endif
#  if __has_builtin(__builtin_amdgcn_s_wait_asynccnt)
#    define HAVE_WAIT_ASYNC 1
#  endif
#endif

// 16-byte global -> LDS copy: async (ASYNCcnt, no VGPR bounce) if available.
// ROCm 7.2 proto: (int4 AS1* src, int4 AS3* dst, imm offset, imm cpol).
__device__ __forceinline__ void async_copy16(const _Float16* g, _Float16* l) {
#ifdef HAVE_ASYNC_LDS
  __attribute__((address_space(1))) v4i_gcc* gp =
      (__attribute__((address_space(1))) v4i_gcc*)(
          (__attribute__((address_space(1))) void*)(g));
  __attribute__((address_space(3))) v4i_gcc* lp =
      (__attribute__((address_space(3))) v4i_gcc*)(
          (__attribute__((address_space(3))) void*)(l));
  __builtin_amdgcn_global_load_async_to_lds_b128(gp, lp, 0, 0);
#else
  *(uint4*)l = *(const uint4*)g;
#endif
}

__device__ __forceinline__ void wait_async0() {
#ifdef HAVE_ASYNC_LDS
#  ifdef HAVE_WAIT_ASYNC
  __builtin_amdgcn_s_wait_asynccnt(0);
#  else
  asm volatile("s_wait_asynccnt 0x0" ::: "memory");
#  endif
#endif
}

__device__ __forceinline__ v8f zero8() {
  v8f z = {0.f, 0.f, 0.f, 0.f, 0.f, 0.f, 0.f, 0.f};
  return z;
}

__device__ __forceinline__ v8f wmma_f16(v16h a, v16h b, v8f c) {
  // D = A(16x32 f16) * B(32x16 f16) + C(16x16 f32)
  return __builtin_amdgcn_wmma_f32_16x16x32_f16(false, a, false, b, (short)0, c,
                                                false, false);
}

// A-matrix fragment (16x32 f16) from row-major LDS tile, stride ldm (halves).
// ISA 7.12.2: lane m=lane&15, half=lane>>4; vgpr v: K=(v&3)*2+(v>>2)*16+8*half.
// Per lane this is two contiguous 16B chunks -> vectorizes to ds_load_b128.
__device__ __forceinline__ v16h load_a(const _Float16* s, int ldm) {
  const int lane = threadIdx.x & 31;
  const int m = lane & 15, hl = lane >> 4;
  v16h a;
#pragma unroll
  for (int e = 0; e < 16; ++e) {
    const int v = e >> 1;
    const int k = ((v & 3) * 2 + (v >> 2) * 16) + 8 * hl + (e & 1);
    a[e] = s[m * ldm + k];
  }
  return a;
}

// B-matrix fragment (32x16 f16) where the LDS tile is stored TRANSPOSED:
// [n][k] with stride ldk. Per lane: two 8-half chunks (k=8*hl, 16+8*hl) -> ds_load_b128 x2.
__device__ __forceinline__ v16h load_bt(const _Float16* s, int ldk) {
  const int lane = threadIdx.x & 31;
  const int n = lane & 15, hl = lane >> 4;
  v16h b;
#pragma unroll
  for (int e = 0; e < 16; ++e) {
    const int k = (e & 7) + 8 * hl + 16 * (e >> 3);
    b[e] = s[n * ldk + k];
  }
  return b;
}

// C/D fragment (16x16 f32) -> LDS row-major, stride ldm floats, scaled by mul.
__device__ __forceinline__ void store_c(const v8f& c, float* s, int ldm, float mul) {
  const int lane = threadIdx.x & 31;
  const int n = lane & 15, hl = lane >> 4;
#pragma unroll
  for (int r = 0; r < 8; ++r) s[(r + 8 * hl) * ldm + n] = c[r] * mul;
}

// ---------------------------------------------------------------------------
// Elementwise / reduction kernels
// ---------------------------------------------------------------------------

__global__ void zero_scales_kernel(float* scales) {
  if (threadIdx.x < 4) scales[threadIdx.x] = 0.f;
}

__global__ void f32_to_f16_kernel(const float* __restrict__ in,
                                  _Float16* __restrict__ out, size_t n) {
  size_t i = (size_t)blockIdx.x * blockDim.x + threadIdx.x;
  if (i < n) out[i] = (_Float16)in[i];
}

__global__ __launch_bounds__(256) void rmsnorm_absmax_kernel(
    const float* __restrict__ x, const float* __restrict__ wgt,
    float* __restrict__ nxf, _Float16* __restrict__ nxh,
    float* __restrict__ absmax_slot) {
  __shared__ float red[256];
  const int row = blockIdx.x;
  const int tid = threadIdx.x;
  const float* xr = x + (size_t)row * Dm;
  float ss = 0.f;
  for (int i = tid; i < Dm; i += 256) { float v = xr[i]; ss += v * v; }
  red[tid] = ss;
  __syncthreads();
  for (int s = 128; s > 0; s >>= 1) {
    if (tid < s) red[tid] += red[tid + s];
    __syncthreads();
  }
  const float rms = rsqrtf(red[0] / (float)Dm + 1e-6f);
  __syncthreads();
  float amax = 0.f;
  for (int i = tid; i < Dm; i += 256) {
    float v = xr[i] * rms * wgt[i];
    nxf[(size_t)row * Dm + i] = v;
    nxh[(size_t)row * Dm + i] = (_Float16)v;
    amax = fmaxf(amax, fabsf(v));
  }
  red[tid] = amax;
  __syncthreads();
  for (int s = 128; s > 0; s >>= 1) {
    if (tid < s) red[tid] = fmaxf(red[tid], red[tid + s]);
    __syncthreads();
  }
  if (tid == 0) atomicMax((unsigned int*)absmax_slot, __float_as_uint(red[0]));
}

__global__ void quantize_f32_kernel(const float* __restrict__ in,
                                    const float* __restrict__ slot,
                                    _Float16* __restrict__ out, size_t n) {
  size_t i = (size_t)blockIdx.x * blockDim.x + threadIdx.x;
  if (i >= n) return;
  const float sc = slot[0] / 127.0f + 1e-8f;
  float q = rintf(in[i] / sc);
  q = fminf(fmaxf(q, -128.f), 127.f);
  out[i] = (_Float16)q;   // integer-valued: exact in f16
}

__global__ void quantize_f16_kernel(const _Float16* __restrict__ in,
                                    const float* __restrict__ slot,
                                    _Float16* __restrict__ out, size_t n) {
  size_t i = (size_t)blockIdx.x * blockDim.x + threadIdx.x;
  if (i >= n) return;
  const float sc = slot[0] / 127.0f + 1e-8f;
  float q = rintf((float)in[i] / sc);
  q = fminf(fmaxf(q, -128.f), 127.f);
  out[i] = (_Float16)q;
}

__global__ __launch_bounds__(256) void swiglu_gate_kernel(
    const _Float16* __restrict__ g1, const _Float16* __restrict__ g2,
    _Float16* __restrict__ out, float* __restrict__ slot, size_t n) {
  __shared__ float red[256];
  size_t i = (size_t)blockIdx.x * 256 + threadIdx.x;
  float am = 0.f;
  if (i < n) {
    float a = (float)g1[i], b = (float)g2[i];
    float v = (a / (1.f + __expf(-a))) * b;
    out[i] = (_Float16)v;
    am = fabsf(v);
  }
  red[threadIdx.x] = am;
  __syncthreads();
  for (int s = 128; s > 0; s >>= 1) {
    if (threadIdx.x < s) red[threadIdx.x] = fmaxf(red[threadIdx.x], red[threadIdx.x + s]);
    __syncthreads();
  }
  if (threadIdx.x == 0) atomicMax((unsigned int*)slot, __float_as_uint(red[0]));
}

__global__ void dwconv_kernel(const float* __restrict__ nx,
                              const float* __restrict__ cw,
                              const float* __restrict__ cb,
                              float* __restrict__ out) {
  size_t i = (size_t)blockIdx.x * 256 + threadIdx.x;
  if (i >= (size_t)NT * Dm) return;
  const int d = (int)(i % Dm);
  const size_t row = i / Dm;
  const int s = (int)(row % Ss);
  float acc = cb[d];
#pragma unroll
  for (int j = 0; j < 5; ++j) {
    const int so = s + j - 2;
    if (so >= 0 && so < Ss)
      acc += nx[(row + (size_t)(j - 2)) * Dm + d] * cw[d * 5 + j];
  }
  out[i] = acc;
}

// ---------------------------------------------------------------------------
// WMMA GEMM: C[M,N] = A[M,K](f16) @ W[N,K](f16)^T, f32 accumulate.
// 128x128 block tile, 8 waves (32M x 64N per wave), K-step 32,
// double-buffered LDS with async-to-LDS staging.
// ---------------------------------------------------------------------------
// epilogue modes:
//  0: outh = acc + bias                                  (in_proj -> qkv f16)
//  1: outf = tanh(acc*sc + bias + exp(decay[n])*r0)      (recurrence -> h_rec)
//  2: outf = acc + bias + r0 + r1 + r2                   (out_proj + residual)
//  3: outh = acc*sc + bias                               (w1 / w2)
//  4: outf = acc*sc + bias + r0                          (w3 + residual)

__global__ __launch_bounds__(256) void gemm_wmma_kernel(
    const _Float16* __restrict__ A, const _Float16* __restrict__ W,
    const float* __restrict__ bias, int M, int N, int K, int mode,
    const float* __restrict__ scale_slot, const float* __restrict__ r0,
    const float* __restrict__ r1, const float* __restrict__ r2,
    const float* __restrict__ decay, float* __restrict__ outf,
    _Float16* __restrict__ outh) {
  __shared__ __align__(16) _Float16 sA[2][128 * 32];
  __shared__ __align__(16) _Float16 sB[2][128 * 32];

  const int tid = threadIdx.x;
  const int w = tid >> 5;
  const int wm = w >> 1;        // 0..3 -> 32 rows each
  const int wn = w & 1;         // 0..1 -> 64 cols each
  const int m0 = blockIdx.y * 128;
  const int n0 = blockIdx.x * 128;
  const int lrow = tid >> 1;    // 0..127
  const int lseg = tid & 1;     // 16 halves each

  auto stage = [&](int buf, int k0) {
    async_copy16(A + (size_t)(m0 + lrow) * K + k0 + lseg * 16,
                 &sA[buf][lrow * 32 + lseg * 16]);
    async_copy16(W + (size_t)(n0 + lrow) * K + k0 + lseg * 16,
                 &sB[buf][lrow * 32 + lseg * 16]);
  };

  v8f acc[2][4];
#pragma unroll
  for (int i = 0; i < 2; ++i)
#pragma unroll
    for (int j = 0; j < 4; ++j) acc[i][j] = zero8();

  stage(0, 0);
  wait_async0();
  __syncthreads();

  int cur = 0;
  for (int k0 = 0; k0 < K; k0 += 32) {
    // overlap: issue async stage of next slab before computing this one
    if (k0 + 32 < K) stage(cur ^ 1, k0 + 32);
    if (k0 + 64 < K) {  // prefetch slab after next into L2 (global_prefetch_b8)
      __builtin_prefetch(A + (size_t)(m0 + lrow) * K + k0 + 64, 0, 1);
      __builtin_prefetch(W + (size_t)(n0 + lrow) * K + k0 + 64, 0, 1);
    }

    v16h af[2], bf[4];
#pragma unroll
    for (int mf = 0; mf < 2; ++mf)
      af[mf] = load_a(&sA[cur][(wm * 32 + mf * 16) * 32], 32);
#pragma unroll
    for (int nf = 0; nf < 4; ++nf)
      bf[nf] = load_bt(&sB[cur][(wn * 64 + nf * 16) * 32], 32);
#pragma unroll
    for (int mf = 0; mf < 2; ++mf)
#pragma unroll
      for (int nf = 0; nf < 4; ++nf)
        acc[mf][nf] = wmma_f16(af[mf], bf[nf], acc[mf][nf]);

    wait_async0();
    __syncthreads();
    cur ^= 1;
  }

  float sc = 1.0f;
  if (scale_slot) sc = scale_slot[0] / 127.0f + 1e-8f;
  const int lane = tid & 31;
  const int nloc = lane & 15, hl = lane >> 4;

#pragma unroll
  for (int mf = 0; mf < 2; ++mf) {
#pragma unroll
    for (int nf = 0; nf < 4; ++nf) {
#pragma unroll
      for (int r = 0; r < 8; ++r) {
        const int row = m0 + wm * 32 + mf * 16 + r + 8 * hl;
        const int col = n0 + wn * 64 + nf * 16 + nloc;
        const size_t idx = (size_t)row * N + col;
        float v = acc[mf][nf][r] * sc + bias[col];
        switch (mode) {
          case 0: outh[idx] = (_Float16)v; break;
          case 1: outf[idx] = tanhf(v + __expf(decay[col]) * r0[idx]); break;
          case 2: outf[idx] = v + r0[idx] + r1[idx] + r2[idx]; break;
          case 3: outh[idx] = (_Float16)v; break;
          case 4: outf[idx] = v + r0[idx]; break;
        }
      }
    }
  }
}

// ---------------------------------------------------------------------------
// Flash attention: workgroup = (b, head, 64-row q block), 4 waves, 16 q rows
// per wave. K staged async row-major; V staged TRANSPOSED ([d][key], stride
// 72 halves) so PV B-fragments are contiguous per lane (ds_load_b128).
// sS padded to stride 68 floats: bank = 4r + 32*hl + n -> conflict-free.
// ---------------------------------------------------------------------------
constexpr int VT_LD = 72;   // sVt stride in halves (16B-aligned rows)
constexpr int SS_LD = 68;   // sS stride in floats

__global__ __launch_bounds__(128) void flash_attn_kernel(
    const _Float16* __restrict__ qkv, _Float16* __restrict__ out) {
  const int qb = blockIdx.x;     // 0..31
  const int head = blockIdx.y;   // 0..15
  const int b = blockIdx.z;      // 0..3

  __shared__ __align__(16) _Float16 sK[64 * 64];
  __shared__ __align__(16) _Float16 sVt[64 * VT_LD];
  __shared__ float    sS[4][16 * SS_LD];
  __shared__ _Float16 sP[4][16 * 64];
  __shared__ float    sCorr[4][16];
  __shared__ float    sL[4][16];

  const int tid = threadIdx.x, w = tid >> 5, lane = tid & 31;
  const int nloc = lane & 15, hl = lane >> 4;
  const size_t base = (size_t)b * Ss * (3 * Dm) + (size_t)head * HDm;
  const int r = tid >> 1, seg = tid & 1;   // staging: 64 rows x 2 half-rows

  // ---- stage Q block (async) into sK, extract A fragments ----------------
  {
    const _Float16* src = qkv + base + (size_t)(qb * 64 + r) * (3 * Dm) + seg * 32;
#pragma unroll
    for (int j = 0; j < 4; ++j)
      async_copy16(src + j * 8, &sK[r * 64 + seg * 32 + j * 8]);
  }
  wait_async0();
  __syncthreads();
  v16h aQ[2];
  aQ[0] = load_a(&sK[(w * 16) * 64], 64);
  aQ[1] = load_a(&sK[(w * 16) * 64 + 32], 64);
  __syncthreads();

  v8f o[4];
#pragma unroll
  for (int dt = 0; dt < 4; ++dt) o[dt] = zero8();
  float mrow = -1e30f, lrow = 0.f;   // lanes 0..15 own one q row each

  for (int kt = 0; kt < Ss / 64; ++kt) {
    // ---- stage K (async, row-major) and V (transposed via VGPRs) ----
    {
      const _Float16* ksrc =
          qkv + base + Dm + (size_t)(kt * 64 + r) * (3 * Dm) + seg * 32;
      const _Float16* vsrc = ksrc + Dm;
#pragma unroll
      for (int j = 0; j < 4; ++j)
        async_copy16(ksrc + j * 8, &sK[r * 64 + seg * 32 + j * 8]);
#pragma unroll
      for (int j = 0; j < 4; ++j) {
        uint4 t = *(const uint4*)(vsrc + j * 8);
        const _Float16* th = (const _Float16*)&t;
#pragma unroll
        for (int u = 0; u < 8; ++u)
          sVt[(seg * 32 + j * 8 + u) * VT_LD + r] = th[u];
      }
    }
    wait_async0();
    __syncthreads();

    // ---- S = Q K^T * (1/sqrt(HD)) ----
#pragma unroll
    for (int nt = 0; nt < 4; ++nt) {
      v8f s = zero8();
      s = wmma_f16(aQ[0], load_bt(&sK[(nt * 16) * 64], 64), s);
      s = wmma_f16(aQ[1], load_bt(&sK[(nt * 16) * 64 + 32], 64), s);
      store_c(s, &sS[w][nt * 16], SS_LD, 0.125f);
    }
    __syncthreads();

    // ---- online softmax (lanes 0..15 own one q row each) ----
    if (lane < 16) {
      float* srow = &sS[w][lane * SS_LD];
      float mx = mrow;
#pragma unroll 8
      for (int j = 0; j < 64; ++j) mx = fmaxf(mx, srow[j]);
      const float corr = __expf(mrow - mx);
      float ls = 0.f;
#pragma unroll 8
      for (int j = 0; j < 64; ++j) {
        const float p = __expf(srow[j] - mx);
        ls += p;
        sP[w][lane * 64 + j] = (_Float16)p;
      }
      lrow = lrow * corr + ls;
      mrow = mx;
      sCorr[w][lane] = corr;
    }
    __syncthreads();

    // ---- O = O*corr + P @ V ----
    float corrv[8];
#pragma unroll
    for (int rr = 0; rr < 8; ++rr) corrv[rr] = sCorr[w][rr + 8 * hl];
    const v16h aP0 = load_a(&sP[w][0], 64);
    const v16h aP1 = load_a(&sP[w][32], 64);
#pragma unroll
    for (int dt = 0; dt < 4; ++dt) {
#pragma unroll
      for (int rr = 0; rr < 8; ++rr) o[dt][rr] *= corrv[rr];
      o[dt] = wmma_f16(aP0, load_bt(&sVt[(dt * 16) * VT_LD], VT_LD), o[dt]);
      o[dt] = wmma_f16(aP1, load_bt(&sVt[(dt * 16) * VT_LD + 32], VT_LD), o[dt]);
    }
    __syncthreads();
  }

  if (lane < 16) sL[w][lane] = lrow;
  __syncthreads();

  const size_t tok0 = (size_t)b * Ss + qb * 64 + w * 16;
#pragma unroll
  for (int rr = 0; rr < 8; ++rr) {
    const float linv = 1.0f / sL[w][rr + 8 * hl];
    const size_t tok = tok0 + rr + 8 * hl;
#pragma unroll
    for (int dt = 0; dt < 4; ++dt)
      out[tok * Dm + (size_t)head * HDm + dt * 16 + nloc] =
          (_Float16)(o[dt][rr] * linv);
  }
}

// ---------------------------------------------------------------------------
// Host-side orchestration
// ---------------------------------------------------------------------------
extern "C" void kernel_launch(void* const* d_in, const int* in_sizes, int n_in,
                              void* d_out, int out_size, void* d_ws, size_t ws_size,
                              hipStream_t stream) {
  const float* x        = (const float*)d_in[0];
  const float* h_prev   = (const float*)d_in[1];
  const float* norm1_w  = (const float*)d_in[2];
  const float* norm2_w  = (const float*)d_in[3];
  const float* decay    = (const float*)d_in[4];
  const float* rec_W    = (const float*)d_in[5];
  const float* rec_b    = (const float*)d_in[6];
  const float* inp_W    = (const float*)d_in[7];
  const float* inp_b    = (const float*)d_in[8];
  const float* out_W    = (const float*)d_in[9];
  const float* out_b    = (const float*)d_in[10];
  const float* conv_w   = (const float*)d_in[11];
  const float* conv_b   = (const float*)d_in[12];
  const float* w1_W     = (const float*)d_in[13];
  const float* w1_b     = (const float*)d_in[14];
  const float* w2_W     = (const float*)d_in[15];
  const float* w2_b     = (const float*)d_in[16];
  const float* w3_W     = (const float*)d_in[17];
  const float* w3_b     = (const float*)d_in[18];

  float* out_x    = (float*)d_out;                       // [NT, D]
  float* out_hrec = (float*)d_out + (size_t)NT * Dm;     // [NT, D]

  // ---- workspace layout ----
  char* ws = (char*)d_ws;
  size_t off = 0;
  auto take = [&](size_t bytes) {
    size_t o = off;
    off += (bytes + 255) & ~(size_t)255;
    return o;
  };
  float*    scales  = (float*)(ws + take(256));                        // absmax[3]
  float*    nx_f32  = (float*)(ws + take((size_t)NT * Dm * 4));
  _Float16* nx_f16  = (_Float16*)(ws + take((size_t)NT * Dm * 2));
  _Float16* q_f16   = (_Float16*)(ws + take((size_t)NT * Dm * 2));
  _Float16* qkv     = (_Float16*)(ws + take((size_t)NT * 3 * Dm * 2));
  _Float16* attn    = (_Float16*)(ws + take((size_t)NT * Dm * 2));
  float*    convo   = (float*)(ws + take((size_t)NT * Dm * 4));
  float*    x2      = (float*)(ws + take((size_t)NT * Dm * 4));
  _Float16* g1h     = (_Float16*)(ws + take((size_t)NT * DFF * 2));
  _Float16* g2h     = (_Float16*)(ws + take((size_t)NT * DFF * 2));
  _Float16* q3      = (_Float16*)(ws + take((size_t)NT * DFF * 2));
  _Float16* rec_Wh  = (_Float16*)(ws + take((size_t)Dm * Dm * 2));
  _Float16* inp_Wh  = (_Float16*)(ws + take((size_t)3 * Dm * Dm * 2));
  _Float16* out_Wh  = (_Float16*)(ws + take((size_t)Dm * Dm * 2));
  _Float16* w1_Wh   = (_Float16*)(ws + take((size_t)DFF * Dm * 2));
  _Float16* w2_Wh   = (_Float16*)(ws + take((size_t)DFF * Dm * 2));
  _Float16* w3_Wh   = (_Float16*)(ws + take((size_t)Dm * DFF * 2));
  (void)ws_size; (void)in_sizes; (void)n_in; (void)out_size;

  const size_t ND  = (size_t)NT * Dm;    // 8.4M
  const size_t NF  = (size_t)NT * DFF;   // 33.6M

  zero_scales_kernel<<<1, 32, 0, stream>>>(scales);

  // weight conversions to f16
  auto cvt = [&](const float* src, _Float16* dst, size_t n) {
    f32_to_f16_kernel<<<(unsigned)((n + 255) / 256), 256, 0, stream>>>(src, dst, n);
  };
  cvt(rec_W, rec_Wh, (size_t)Dm * Dm);
  cvt(inp_W, inp_Wh, (size_t)3 * Dm * Dm);
  cvt(out_W, out_Wh, (size_t)Dm * Dm);
  cvt(w1_W,  w1_Wh,  (size_t)DFF * Dm);
  cvt(w2_W,  w2_Wh,  (size_t)DFF * Dm);
  cvt(w3_W,  w3_Wh,  (size_t)Dm * DFF);

  // nx = rmsnorm(x, norm1_w); absmax -> scales[0]
  rmsnorm_absmax_kernel<<<NT, 256, 0, stream>>>(x, norm1_w, nx_f32, nx_f16, &scales[0]);
  // q1 = fake-quant(nx) as int-valued f16
  quantize_f32_kernel<<<(unsigned)(ND / 256), 256, 0, stream>>>(nx_f32, &scales[0], q_f16, ND);

  // h_rec = tanh(q1 @ rec_W^T * sc + rec_b + exp(decay)*h_prev) -> out_hrec
  gemm_wmma_kernel<<<dim3(Dm / 128, NT / 128), 256, 0, stream>>>(
      q_f16, rec_Wh, rec_b, NT, Dm, Dm, 1, &scales[0],
      h_prev, nullptr, nullptr, decay, out_hrec, nullptr);

  // qkv = nx @ in_proj_w^T + in_proj_b  (f16)
  gemm_wmma_kernel<<<dim3(3 * Dm / 128, NT / 128), 256, 0, stream>>>(
      nx_f16, inp_Wh, inp_b, NT, 3 * Dm, Dm, 0, nullptr,
      nullptr, nullptr, nullptr, nullptr, nullptr, qkv);

  // depthwise conv on nx
  dwconv_kernel<<<(unsigned)(ND / 256), 256, 0, stream>>>(nx_f32, conv_w, conv_b, convo);

  // flash attention -> attn (f16)
  flash_attn_kernel<<<dim3(Ss / 64, Hh, Bb), 128, 0, stream>>>(qkv, attn);

  // x2 = x + h_rec + (attn @ out_w^T + out_b) + conv
  gemm_wmma_kernel<<<dim3(Dm / 128, NT / 128), 256, 0, stream>>>(
      attn, out_Wh, out_b, NT, Dm, Dm, 2, nullptr,
      x, out_hrec, convo, nullptr, x2, nullptr);

  // nx2 = rmsnorm(x2, norm2_w); absmax -> scales[1]
  rmsnorm_absmax_kernel<<<NT, 256, 0, stream>>>(x2, norm2_w, nx_f32, nx_f16, &scales[1]);
  quantize_f32_kernel<<<(unsigned)(ND / 256), 256, 0, stream>>>(nx_f32, &scales[1], q_f16, ND);

  // g1 = q2 @ w1^T * sc + b1 ; g2 = q2 @ w2^T * sc + b2   (f16)
  gemm_wmma_kernel<<<dim3(DFF / 128, NT / 128), 256, 0, stream>>>(
      q_f16, w1_Wh, w1_b, NT, DFF, Dm, 3, &scales[1],
      nullptr, nullptr, nullptr, nullptr, nullptr, g1h);
  gemm_wmma_kernel<<<dim3(DFF / 128, NT / 128), 256, 0, stream>>>(
      q_f16, w2_Wh, w2_b, NT, DFF, Dm, 3, &scales[1],
      nullptr, nullptr, nullptr, nullptr, nullptr, g2h);

  // gate = silu(g1)*g2 ; absmax -> scales[2] ; quantize -> q3
  swiglu_gate_kernel<<<(unsigned)(NF / 256), 256, 0, stream>>>(g1h, g2h, g1h, &scales[2], NF);
  quantize_f16_kernel<<<(unsigned)(NF / 256), 256, 0, stream>>>(g1h, &scales[2], q3, NF);

  // out_x = x2 + (q3 @ w3^T * sc + b3)
  gemm_wmma_kernel<<<dim3(Dm / 128, NT / 128), 256, 0, stream>>>(
      q3, w3_Wh, w3_b, NT, Dm, DFF, 4, &scales[2],
      x2, nullptr, nullptr, nullptr, out_x, nullptr);
}
